// TransformerEncoder_89309549953251
// MI455X (gfx1250) — matrix operand order repaired and checked by reading, not compiled, not run on hardware
//
#include <hip/hip_runtime.h>
#include <hip/hip_bf16.h>
#include <math.h>

// ---------------- problem constants ----------------
constexpr int kEMB   = 256;
constexpr int kHEADS = 8;
constexpr int kDH    = 32;
constexpr int kSEQ   = 8192;
constexpr int kBATCH = 4;
constexpr int kFF    = 1024;
constexpr int kDEPTH = 4;
constexpr int kNTOK  = kBATCH * kSEQ;          // 32768 rows
constexpr float kEPS = 1e-5f;
constexpr float kSCALE = 0.42044820762685725f; // 32^-0.25

typedef __attribute__((ext_vector_type(16))) _Float16 v16h;
typedef __attribute__((ext_vector_type(8)))  _Float16 v8h;
typedef __attribute__((ext_vector_type(8)))  float    v8f;
typedef __attribute__((ext_vector_type(4)))  int      v4i;

union AFrag { v16h v; v8h p[2]; };

// gfx1250 async global->LDS staging (ASYNCcnt-tracked), if toolchain exposes it
#if defined(__has_builtin)
#if __has_builtin(__builtin_amdgcn_global_load_async_to_lds_b128) && \
    __has_builtin(__builtin_amdgcn_s_wait_asynccnt)
#define USE_ASYNC_LDS 1
#endif
#endif

#define AS_GLOBAL __attribute__((address_space(1)))
#define AS_SHARED __attribute__((address_space(3)))

// ---------------- weight cast + transpose:  W[K][N] f32 -> Wt[N][K] f16 ----------------
__global__ __launch_bounds__(256)
void transpose_cast_kernel(const float* __restrict__ W, _Float16* __restrict__ Wt,
                           int K, int Nn)
{
    int i = blockIdx.x * 256 + threadIdx.x;       // index into Wt, [Nn][K]
    if (i >= K * Nn) return;
    int n = i / K;
    int k = i - n * K;
    Wt[i] = (_Float16)W[(size_t)k * Nn + n];
}

// ---------------- layernorm + f16 cast (wave per row) ----------------
__global__ __launch_bounds__(256)
void layernorm_cast_kernel(const float* __restrict__ x, const float* __restrict__ g,
                           const float* __restrict__ b, _Float16* __restrict__ h)
{
    int row  = blockIdx.x * 8 + (threadIdx.x >> 5);
    int lane = threadIdx.x & 31;
    const float* xr = x + (size_t)row * kEMB;

    float v[8];
    float s = 0.f;
#pragma unroll
    for (int i = 0; i < 8; ++i) { v[i] = xr[lane + i * 32]; s += v[i]; }
#pragma unroll
    for (int o = 16; o > 0; o >>= 1) s += __shfl_xor(s, o, 32);
    float mu = s * (1.0f / kEMB);

    float q = 0.f;
#pragma unroll
    for (int i = 0; i < 8; ++i) { float d = v[i] - mu; q += d * d; }
#pragma unroll
    for (int o = 16; o > 0; o >>= 1) q += __shfl_xor(q, o, 32);
    float rs = rsqrtf(q * (1.0f / kEMB) + kEPS);

    _Float16* hr = h + (size_t)row * kEMB;
#pragma unroll
    for (int i = 0; i < 8; ++i) {
        int c = lane + i * 32;
        hr[c] = (_Float16)((v[i] - mu) * rs * g[c] + b[c]);
    }
}

// ---------------- WMMA GEMM:  C[M][Nn] = A[M][K](f16) * Bt[Nn][K](f16)^T  ----------------
// block: 256 threads = 8 waves; wave computes 32x64; block computes 256x64.
// B tile (32K x 64N) double-buffered through LDS with async global->LDS copies.
template<bool BIAS, bool GELU, bool RESID, bool OUT16>
__global__ __launch_bounds__(256)
void gemm_wmma_kernel(const _Float16* __restrict__ A, const _Float16* __restrict__ Bt,
                      const float* __restrict__ bias, float* __restrict__ Cf,
                      _Float16* __restrict__ Ch, int M, int Nn, int K)
{
    __shared__ __align__(16) _Float16 lb[2][64 * 32];  // double-buffered B tile (8 KB)

    const int lane = threadIdx.x & 31;
    const int wave = threadIdx.x >> 5;
    const int m0 = blockIdx.x * 256 + wave * 32;
    const int n0 = blockIdx.y * 64;
    const int hi  = lane >> 4;      // 0 or 1
    const int l16 = lane & 15;

    v8f zero8 = {0.f, 0.f, 0.f, 0.f, 0.f, 0.f, 0.f, 0.f};
    v8f acc0[4] = {zero8, zero8, zero8, zero8};
    v8f acc1[4] = {zero8, zero8, zero8, zero8};

    const _Float16* Arow0 = A + (size_t)(m0 + l16) * K;
    const _Float16* Arow1 = A + (size_t)(m0 + 16 + l16) * K;

    const int cl = threadIdx.x >> 2;          // 0..63  (cooperative B stage)
    const int kq = (threadIdx.x & 3) * 8;     // 0,8,16,24
    const _Float16* bsrc = Bt + (size_t)(n0 + cl) * K + kq;

    auto stage = [&](int buf, int kk) {
#if defined(USE_ASYNC_LDS)
        __builtin_amdgcn_global_load_async_to_lds_b128(
            (AS_GLOBAL v4i*)(bsrc + kk),
            (AS_SHARED v4i*)&lb[buf][cl * 32 + kq],
            0, 0);
#else
        *(v8h*)&lb[buf][cl * 32 + kq] = *(const v8h*)(bsrc + kk);
#endif
    };

    auto mma_tile = [&](int kk, int ib) {
        AFrag a0, a1;
        a0.p[0] = *(const v8h*)(Arow0 + kk + hi * 8);
        a0.p[1] = *(const v8h*)(Arow0 + kk + hi * 8 + 16);
        a1.p[0] = *(const v8h*)(Arow1 + kk + hi * 8);
        a1.p[1] = *(const v8h*)(Arow1 + kk + hi * 8 + 16);
#pragma unroll
        for (int j = 0; j < 4; ++j) {
            AFrag bf;
            int c  = (j * 16 + l16) * 32;
            int kb = hi * 8;
            bf.p[0] = *(const v8h*)&lb[ib][c + kb];
            bf.p[1] = *(const v8h*)&lb[ib][c + kb + 16];
            acc0[j] = __builtin_amdgcn_wmma_f32_16x16x32_f16(
                false, a0.v, false, bf.v, (short)0, acc0[j], false, false);
            acc1[j] = __builtin_amdgcn_wmma_f32_16x16x32_f16(
                false, a1.v, false, bf.v, (short)0, acc1[j], false, false);
        }
    };

    stage(0, 0);   // prologue: first B tile in flight

    // steady state: branch-free body; last iteration peeled below
    int ib = 0;
    int kk = 0;
    for (; kk + 32 < K; kk += 32, ib ^= 1) {
        stage(ib ^ 1, kk + 32);               // next tile overlaps this tile's math
#if defined(USE_ASYNC_LDS)
        __builtin_amdgcn_s_wait_asynccnt(1);  // in-order: current tile done
#endif
        __syncthreads();                      // publish current B tile
        __builtin_prefetch(Arow0 + kk + 32 + hi * 8, 0, 3);
        __builtin_prefetch(Arow1 + kk + 32 + hi * 8, 0, 3);
        mma_tile(kk, ib);
        __syncthreads();                      // all waves done reading lb[ib]
    }
    // peeled final K-step
#if defined(USE_ASYNC_LDS)
    __builtin_amdgcn_s_wait_asynccnt(0);
#endif
    __syncthreads();
    mma_tile(kk, ib);

    // epilogue: bias / gelu / residual / f16 downcast
#pragma unroll
    for (int half = 0; half < 2; ++half) {
        const v8f* accp = half ? acc1 : acc0;
        int rbase = m0 + half * 16 + hi * 8;
#pragma unroll
        for (int j = 0; j < 4; ++j) {
            int col = n0 + j * 16 + l16;
#pragma unroll
            for (int r = 0; r < 8; ++r) {
                int row = rbase + r;
                float v = accp[j][r];
                if (BIAS) v += bias[col];
                if (GELU) v = 0.5f * v * (1.0f + erff(v * 0.70710678118654752f));
                size_t idx = (size_t)row * Nn + col;
                if (RESID) { v += Cf[idx]; Cf[idx] = v; }
                else if (!OUT16) { Cf[idx] = v; }
                if (OUT16) Ch[idx] = (_Float16)v;
            }
        }
    }
}

// ---------------- q softmax over feature dim (wave per head-group) ----------------
__global__ __launch_bounds__(256)
void q_softmax_kernel(float* __restrict__ Q)
{
    int row  = blockIdx.x;
    int head = threadIdx.x >> 5;
    int lane = threadIdx.x & 31;
    float* p = Q + (size_t)row * kEMB + head * kDH;
    float v = p[lane] * kSCALE;
    float m = v;
#pragma unroll
    for (int o = 16; o > 0; o >>= 1) m = fmaxf(m, __shfl_xor(m, o, 32));
    float e = __expf(v - m);
    float s = e;
#pragma unroll
    for (int o = 16; o > 0; o >>= 1) s += __shfl_xor(s, o, 32);
    p[lane] = e / s;
}

// ---------------- k softmax over sequence dim (block per (batch,col)) ----------------
__global__ __launch_bounds__(256)
void k_softmax_kernel(float* __restrict__ K)
{
    int b   = blockIdx.x >> 8;
    int col = blockIdx.x & 255;
    float* base = K + (size_t)b * kSEQ * kEMB + col;
    __shared__ float red[256];
    int t = threadIdx.x;

    float m = -1e30f;
    for (int n = t; n < kSEQ; n += 256) m = fmaxf(m, base[(size_t)n * kEMB] * kSCALE);
    red[t] = m; __syncthreads();
    for (int o = 128; o > 0; o >>= 1) { if (t < o) red[t] = fmaxf(red[t], red[t + o]); __syncthreads(); }
    m = red[0]; __syncthreads();

    float s = 0.f;
    for (int n = t; n < kSEQ; n += 256) s += __expf(base[(size_t)n * kEMB] * kSCALE - m);
    red[t] = s; __syncthreads();
    for (int o = 128; o > 0; o >>= 1) { if (t < o) red[t] += red[t + o]; __syncthreads(); }
    float inv = 1.0f / red[0];

    for (int n = t; n < kSEQ; n += 256)
        base[(size_t)n * kEMB] = __expf(base[(size_t)n * kEMB] * kSCALE - m) * inv;
}

// ---------------- context[b,h] = K^T V  (32x32 per head, reduce over seq) ----------------
__global__ __launch_bounds__(256)
void context_kernel(const float* __restrict__ K, const float* __restrict__ V,
                    float* __restrict__ ctx)
{
    int bh = blockIdx.x;                 // 0..31
    int b = bh >> 3, h = bh & 7;
    __shared__ float Ks[8][32], Vs[8][32];
    int t = threadIdx.x;
    int r = t >> 5, c = t & 31;
    float acc[4] = {0.f, 0.f, 0.f, 0.f};
    const size_t base = (size_t)b * kSEQ * kEMB + (size_t)h * kDH;

    for (int n0 = 0; n0 < kSEQ; n0 += 8) {
        __syncthreads();
        Ks[r][c] = K[base + (size_t)(n0 + r) * kEMB + c];
        Vs[r][c] = V[base + (size_t)(n0 + r) * kEMB + c];
        __syncthreads();
#pragma unroll
        for (int p = 0; p < 4; ++p) {
            int idx = t + p * 256;
            int d = idx >> 5, e = idx & 31;
            float a = 0.f;
#pragma unroll
            for (int rr = 0; rr < 8; ++rr) a += Ks[rr][d] * Vs[rr][e];
            acc[p] += a;
        }
    }
#pragma unroll
    for (int p = 0; p < 4; ++p) ctx[(size_t)bh * 1024 + t + p * 256] = acc[p];
}

// ---------------- out = Q @ ctx  (per head), write f16 ----------------
__global__ __launch_bounds__(256)
void attn_out_kernel(const float* __restrict__ Q, const float* __restrict__ ctx,
                     _Float16* __restrict__ ao)
{
    __shared__ float cs[kHEADS][kDH][kDH];   // 32 KB, all heads of this batch
    int t = threadIdx.x;
    int b = (blockIdx.x * 32) / kSEQ;
    for (int i = t; i < kHEADS * kDH * kDH; i += 256)
        ((float*)cs)[i] = ctx[(size_t)b * kHEADS * kDH * kDH + i];
    __syncthreads();

    int h = t >> 5, e = t & 31;
    for (int rr = 0; rr < 32; ++rr) {
        int row = blockIdx.x * 32 + rr;
        const float* q = Q + (size_t)row * kEMB + h * kDH;
        float v = 0.f;
#pragma unroll
        for (int d = 0; d < kDH; ++d) v += q[d] * cs[h][d][e];
        ao[(size_t)row * kEMB + t] = (_Float16)v;
    }
}

// ---------------- host orchestration ----------------
extern "C" void kernel_launch(void* const* d_in, const int* in_sizes, int n_in,
                              void* d_out, int out_size, void* d_ws, size_t ws_size,
                              hipStream_t stream)
{
    (void)in_sizes; (void)n_in; (void)out_size; (void)ws_size;

    const float* x_in  = (const float*)d_in[0];
    const float* Wq    = (const float*)d_in[1];
    const float* Wk    = (const float*)d_in[2];
    const float* Wv    = (const float*)d_in[3];
    const float* Wo    = (const float*)d_in[4];
    const float* bo    = (const float*)d_in[5];
    const float* ln1g  = (const float*)d_in[6];
    const float* ln1b  = (const float*)d_in[7];
    const float* W1    = (const float*)d_in[8];
    const float* b1    = (const float*)d_in[9];
    const float* W2    = (const float*)d_in[10];
    const float* b2    = (const float*)d_in[11];
    const float* ln2g  = (const float*)d_in[12];
    const float* ln2b  = (const float*)d_in[13];

    float* x = (float*)d_out;   // activations live in d_out; residuals in place
    (void)hipMemcpyAsync(x, x_in, (size_t)kNTOK * kEMB * sizeof(float),
                         hipMemcpyDeviceToDevice, stream);

    // ---- workspace carve-up ----
    char* ws = (char*)d_ws;
    size_t off = 0;
    auto take = [&](size_t bytes) -> char* {
        char* p = ws + off;
        off += (bytes + 255) & ~(size_t)255;
        return p;
    };
    _Float16* h16 = (_Float16*)take((size_t)kNTOK * kEMB * 2);   // 16 MB
    float*    Qf  = (float*)take((size_t)kNTOK * kEMB * 4);      // 32 MB
    float*    Kf  = (float*)take((size_t)kNTOK * kEMB * 4);      // 32 MB (Qf+Kf contiguous)
    float*    Vf  = (float*)take((size_t)kNTOK * kEMB * 4);      // 32 MB
    float*    ctx = (float*)take((size_t)kBATCH * kHEADS * kDH * kDH * 4);
    _Float16* wt[kDEPTH][6];
    for (int i = 0; i < kDEPTH; ++i) {
        for (int j = 0; j < 4; ++j) wt[i][j] = (_Float16*)take((size_t)kEMB * kEMB * 2);
        wt[i][4] = (_Float16*)take((size_t)kEMB * kFF * 2);
        wt[i][5] = (_Float16*)take((size_t)kFF * kEMB * 2);
    }
    _Float16* ao16  = (_Float16*)Kf;   // alias: ao consumed before h1 produced
    _Float16* h1_16 = (_Float16*)Qf;   // alias: spans Qf+Kf (64 MB)

    // ---- cast + transpose weights to f16 [out][in] ----
    for (int i = 0; i < kDEPTH; ++i) {
        transpose_cast_kernel<<<(kEMB * kEMB) / 256, 256, 0, stream>>>(
            Wq + (size_t)i * kEMB * kEMB, wt[i][0], kEMB, kEMB);
        transpose_cast_kernel<<<(kEMB * kEMB) / 256, 256, 0, stream>>>(
            Wk + (size_t)i * kEMB * kEMB, wt[i][1], kEMB, kEMB);
        transpose_cast_kernel<<<(kEMB * kEMB) / 256, 256, 0, stream>>>(
            Wv + (size_t)i * kEMB * kEMB, wt[i][2], kEMB, kEMB);
        transpose_cast_kernel<<<(kEMB * kEMB) / 256, 256, 0, stream>>>(
            Wo + (size_t)i * kEMB * kEMB, wt[i][3], kEMB, kEMB);
        transpose_cast_kernel<<<(kEMB * kFF) / 256, 256, 0, stream>>>(
            W1 + (size_t)i * kEMB * kFF, wt[i][4], kEMB, kFF);
        transpose_cast_kernel<<<(kFF * kEMB) / 256, 256, 0, stream>>>(
            W2 + (size_t)i * kFF * kEMB, wt[i][5], kFF, kEMB);
    }

    dim3 gEMB(kNTOK / 256, kEMB / 64);
    dim3 gFF (kNTOK / 256, kFF  / 64);

    for (int i = 0; i < kDEPTH; ++i) {
        // LN1 -> h16
        layernorm_cast_kernel<<<kNTOK / 8, 256, 0, stream>>>(
            x, ln1g + (size_t)i * kEMB, ln1b + (size_t)i * kEMB, h16);
        // Q,K,V GEMMs
        gemm_wmma_kernel<false,false,false,false><<<gEMB, 256, 0, stream>>>(
            h16, wt[i][0], nullptr, Qf, nullptr, kNTOK, kEMB, kEMB);
        gemm_wmma_kernel<false,false,false,false><<<gEMB, 256, 0, stream>>>(
            h16, wt[i][1], nullptr, Kf, nullptr, kNTOK, kEMB, kEMB);
        gemm_wmma_kernel<false,false,false,false><<<gEMB, 256, 0, stream>>>(
            h16, wt[i][2], nullptr, Vf, nullptr, kNTOK, kEMB, kEMB);
        // softmaxes
        q_softmax_kernel<<<kNTOK, 256, 0, stream>>>(Qf);
        k_softmax_kernel<<<kBATCH * kEMB, 256, 0, stream>>>(Kf);
        // context + attention out
        context_kernel<<<kBATCH * kHEADS, 256, 0, stream>>>(Kf, Vf, ctx);
        attn_out_kernel<<<kNTOK / 32, 256, 0, stream>>>(Qf, ctx, ao16);
        // out-proj + bias + residual into x
        gemm_wmma_kernel<true,false,true,false><<<gEMB, 256, 0, stream>>>(
            ao16, wt[i][3], bo + (size_t)i * kEMB, x, nullptr, kNTOK, kEMB, kEMB);
        // LN2 -> h16
        layernorm_cast_kernel<<<kNTOK / 8, 256, 0, stream>>>(
            x, ln2g + (size_t)i * kEMB, ln2b + (size_t)i * kEMB, h16);
        // FFN up + gelu -> f16
        gemm_wmma_kernel<true,true,false,true><<<gFF, 256, 0, stream>>>(
            h16, wt[i][4], b1 + (size_t)i * kFF, nullptr, h1_16, kNTOK, kFF, kEMB);
        // FFN down + bias + residual into x
        gemm_wmma_kernel<true,false,true,false><<<gEMB, 256, 0, stream>>>(
            h1_16, wt[i][5], b2 + (size_t)i * kEMB, x, nullptr, kNTOK, kEMB, kFF);
    }
}
